// GraphEmbeddings_57166014710439
// MI455X (gfx1250) — matrix operand order repaired and checked by reading, not compiled, not run on hardware
//
#include <hip/hip_runtime.h>
#include <hip/hip_bf16.h>
#include <math.h>

typedef _Float16 f16;
typedef __attribute__((ext_vector_type(16))) _Float16 v16h;
typedef __attribute__((ext_vector_type(8)))  _Float16 v8h;
typedef __attribute__((ext_vector_type(4)))  _Float16 v4h;
typedef __attribute__((ext_vector_type(8)))  float    v8f;
typedef __attribute__((ext_vector_type(4)))  float    v4f;

#define NN 50000
#define EE 800000
#define CC 64
#define HHD 2
#define BBG 500
#define LLS 100
#define TOTE (EE + NN)
#define BPAD 512
#define NEG_SLOPE 0.2f
#define BN_EPS 1e-5f
#define LOG2E 1.44269504088896340736f

// ---------------- workspace layout (byte offsets, 64B aligned) ----------------
constexpr size_t OFF_SUME   = 0;                                   // 1 f32 (padded)
constexpr size_t OFF_AMAX   = 64;                                  // NN*2 u32 (ordered-float keys)
constexpr size_t OFF_DENOM  = OFF_AMAX  + (size_t)NN * HHD * 4;    // NN*2 f32
constexpr size_t OFF_T1     = OFF_DENOM + (size_t)NN * HHD * 4;    // NN*2 f32 (sum ex*x_src)
constexpr size_t OFF_BNSUM  = OFF_T1    + (size_t)NN * HHD * 4;    // 64 f32
constexpr size_t OFF_BNSQ   = OFF_BNSUM + 256;
constexpr size_t OFF_SCALE  = OFF_BNSQ  + 256;
constexpr size_t OFF_SHIFT  = OFF_SCALE + 256;
constexpr size_t OFF_BIAS   = OFF_SHIFT + 256;                     // 256 f32 (b_ih+b_hh)
constexpr size_t OFF_HDENSE = OFF_BIAS  + 1024;                    // BPAD*LLS*CC f16 (zero-padded)
constexpr size_t OFF_WIH    = OFF_HDENSE + (size_t)BPAD * LLS * CC * 2;
constexpr size_t OFF_WHH    = OFF_WIH   + (size_t)(4 * CC) * CC * 2;
constexpr size_t OFF_ALPHA  = OFF_WHH   + (size_t)(4 * CC) * CC * 2;   // TOTE*2 f32
constexpr size_t OFF_HRAW   = OFF_ALPHA + (size_t)TOTE * HHD * 4;      // NN*64 f32
constexpr size_t ZERO_BYTES = OFF_WIH;    // zero everything through hdense (incl. pad rows)

// ---------------- helpers ----------------
__device__ __forceinline__ unsigned encOrd(float f) {
  unsigned b = __float_as_uint(f);
  return (b & 0x80000000u) ? ~b : (b | 0x80000000u);
}
__device__ __forceinline__ float decOrd(unsigned u) {
  return __uint_as_float((u & 0x80000000u) ? (u ^ 0x80000000u) : ~u);
}
// Branchless hardware-transcendental activations: v_exp_f32 + v_rcp_f32.
// sigm(x) = 1/(1+2^(-x*log2e)); saturates correctly (rcp(inf)=0).
__device__ __forceinline__ float fsigm(float x) {
  return __builtin_amdgcn_rcpf(1.f + __builtin_amdgcn_exp2f(-x * LOG2E));
}
// tanh(x) = 2/(1+2^(-2x*log2e)) - 1
__device__ __forceinline__ float ftanh(float x) {
  return 2.f * __builtin_amdgcn_rcpf(1.f + __builtin_amdgcn_exp2f(-x * (2.f * LOG2E))) - 1.f;
}
__device__ __forceinline__ float fexp(float x) {
  return __builtin_amdgcn_exp2f(x * LOG2E);
}

// ---------------- kernels ----------------
__global__ void k_zero(uint4* p, size_t n) {
  size_t i = (size_t)blockIdx.x * blockDim.x + threadIdx.x;
  size_t stride = (size_t)gridDim.x * blockDim.x;
  uint4 z = {0u, 0u, 0u, 0u};
  for (; i < n; i += stride) p[i] = z;
}

__global__ void __launch_bounds__(256) k_edge_sum(const float* __restrict__ ea, float* sum) {
  __shared__ float s;
  if (threadIdx.x == 0) s = 0.f;
  __syncthreads();
  int i = blockIdx.x * 256 + threadIdx.x;
  if (i < EE) atomicAdd(&s, ea[i]);
  __syncthreads();
  if (threadIdx.x == 0) atomicAdd(sum, s);
}

// Pass 1: per-edge attention logits + segment max (ordered-uint atomicMax)
__global__ void __launch_bounds__(256)
k_edge_alpha(const int* __restrict__ ei, const float* __restrict__ eattr,
             const float* __restrict__ x,
             const float* __restrict__ Wl, const float* __restrict__ bl,
             const float* __restrict__ Wr, const float* __restrict__ br,
             const float* __restrict__ We, const float* __restrict__ att,
             const float* __restrict__ sumE,
             float* __restrict__ alphaOut, unsigned* __restrict__ amax) {
  __shared__ float P[768];   // Wl | bl | Wr | br | We | att
  for (int i = threadIdx.x; i < 768; i += 256) {
    int a = i >> 7, c = i & 127;
    float v;
    switch (a) {
      case 0: v = Wl[c]; break;  case 1: v = bl[c]; break;
      case 2: v = Wr[c]; break;  case 3: v = br[c]; break;
      case 4: v = We[c]; break;  default: v = att[c]; break;
    }
    P[i] = v;
  }
  __syncthreads();
  int e = blockIdx.x * 256 + threadIdx.x;
  if (e >= TOTE) return;
  int s, d; float ea;
  if (e < EE) { s = ei[e]; d = ei[EE + e]; ea = eattr[e]; }
  else        { s = d = e - EE; ea = sumE[0] * (1.0f / EE); }   // self-loop, fill='mean'
  float xs = x[s], xd = x[d];
  #pragma unroll
  for (int h = 0; h < 2; ++h) {
    float acc = 0.f;
    #pragma unroll 8
    for (int c = 0; c < 64; ++c) {
      int k = h * 64 + c;
      float v = xs * P[k] + P[128 + k] + xd * P[256 + k] + P[384 + k] + ea * P[512 + k];
      v = (v > 0.f) ? v : NEG_SLOPE * v;   // leaky_relu
      acc += v * P[640 + k];
    }
    alphaOut[(size_t)e * 2 + h] = acc;
    atomicMax(&amax[(size_t)d * 2 + h], encOrd(acc));
  }
}

// Pass 2: ex = exp(alpha - max); accumulate denom and ex*x_src per dst node
__global__ void __launch_bounds__(256)
k_edge_soft(const int* __restrict__ ei, const float* __restrict__ x,
            const float* __restrict__ alpha, const unsigned* __restrict__ amax,
            float* __restrict__ denom, float* __restrict__ T1) {
  int e = blockIdx.x * 256 + threadIdx.x;
  if (e >= TOTE) return;
  int s, d;
  if (e < EE) { s = ei[e]; d = ei[EE + e]; }
  else        { s = d = e - EE; }
  float xs = x[s];
  #pragma unroll
  for (int h = 0; h < 2; ++h) {
    float al = alpha[(size_t)e * 2 + h];
    float mx = decOrd(amax[(size_t)d * 2 + h]);
    float ex = fexp(al - mx);     // alpha <= max, so arg <= 0: no overflow path needed
    atomicAdd(&denom[(size_t)d * 2 + h], ex);
    atomicAdd(&T1[(size_t)d * 2 + h], ex * xs);
  }
}

// Node pass: GAT output via rank-1 reconstruction, head-mean, + BN stats
__global__ void __launch_bounds__(256)
k_node(const float* __restrict__ denom, const float* __restrict__ T1,
       const float* __restrict__ Wl, const float* __restrict__ bl,
       const float* __restrict__ gbias,
       float* __restrict__ hraw, float* __restrict__ bnsum, float* __restrict__ bnsq) {
  __shared__ float ls[64], lsq[64];
  int tid = threadIdx.x;
  if (tid < 64) { ls[tid] = 0.f; lsq[tid] = 0.f; }
  __syncthreads();
  int id = blockIdx.x * 256 + tid;     // exactly NN*64 threads total
  int n = id >> 6, c = id & 63;
  float acc = 0.f;
  #pragma unroll
  for (int h = 0; h < 2; ++h) {
    float dn = denom[(size_t)n * 2 + h];
    float inv = 1.f / (dn + 1e-16f);
    float S1 = T1[(size_t)n * 2 + h] * inv;   // sum(a * x_src)
    float S0 = dn * inv;                      // sum(a)
    acc += Wl[h * 64 + c] * S1 + bl[h * 64 + c] * S0;
  }
  float hv = 0.5f * acc + gbias[c];           // head mean + bias
  hraw[id] = hv;
  atomicAdd(&ls[c], hv);
  atomicAdd(&lsq[c], hv * hv);
  __syncthreads();
  if (tid < 64) { atomicAdd(&bnsum[tid], ls[tid]); atomicAdd(&bnsq[tid], lsq[tid]); }
}

__global__ void k_bnstats(const float* __restrict__ bnsum, const float* __restrict__ bnsq,
                          const float* __restrict__ gamma, const float* __restrict__ beta,
                          float* __restrict__ scale, float* __restrict__ shift) {
  int c = threadIdx.x;
  if (c >= 64) return;
  float mu = bnsum[c] * (1.0f / NN);
  float var = bnsq[c] * (1.0f / NN) - mu * mu;   // biased batch var
  float s = gamma[c] * rsqrtf(var + BN_EPS);
  scale[c] = s;
  shift[c] = beta[c] - mu * s;
}

// BN apply + ReLU + f32->f16 into dense [BPAD, LLS, CC] (rows >= NN stay zero)
__global__ void __launch_bounds__(256)
k_bnapply(const float* __restrict__ hraw, const float* __restrict__ scale,
          const float* __restrict__ shift, f16* __restrict__ hdense) {
  int id = blockIdx.x * 256 + threadIdx.x;
  if (id >= NN * 64) return;
  int c = id & 63;
  float v = fmaxf(0.f, hraw[id] * scale[c] + shift[c]);
  hdense[id] = (f16)v;    // node n == b*LLS + t already (sorted equal-size graphs)
}

// Convert LSTM weights to f16, combine biases
__global__ void __launch_bounds__(256)
k_prep(const float* __restrict__ Wih, const float* __restrict__ Whh,
       const float* __restrict__ bih, const float* __restrict__ bhh,
       f16* __restrict__ wihH, f16* __restrict__ whhH, float* __restrict__ bias) {
  int i = blockIdx.x * 256 + threadIdx.x;
  if (i < 4 * CC * CC) { wihH[i] = (f16)Wih[i]; whhH[i] = (f16)Whh[i]; }
  if (i < 4 * CC) bias[i] = bih[i] + bhh[i];
}

// Persistent LSTM: each block owns 16 independent sequences, runs all 100 steps.
// gates[16,256] = x_t[16,64] @ W_ih^T + h[16,64] @ W_hh^T via v_wmma_f32_16x16x32_f16.
// Weights stay resident in VGPRs as pre-swizzled B-operands for the whole kernel.
__global__ void __launch_bounds__(256)
k_lstm(const f16* __restrict__ hdense, const f16* __restrict__ wih,
       const f16* __restrict__ whh, const float* __restrict__ biasG,
       float* __restrict__ out) {
  __shared__ float gatesLds[16][256];
  __shared__ f16   hbuf[16][64];
  __shared__ float biasLds[256];
  const int tid  = threadIdx.x;
  const int lane = tid & 31;
  const int wave = tid >> 5;           // 8 waves, 2 N-tiles each
  const int rb   = blockIdx.x;         // graphs rb*16 .. rb*16+15

  biasLds[tid] = biasG[tid];
  for (int i = tid; i < 16 * 64; i += 256) (&hbuf[0][0])[i] = (f16)0.f;

  // B-operand layout (32x16 f16): lanes 0-15 hold K=0..15 of column lane,
  // lanes 16-31 hold K=16..31 -> 16 contiguous K-halves of W row n.
  const int nloc = lane & 15;
  const int kb   = (lane >> 4) * 16;
  v16h bih[2][2], bhh[2][2];
  #pragma unroll
  for (int jj = 0; jj < 2; ++jj) {
    int n = (wave * 2 + jj) * 16 + nloc;     // gate column = W row
    #pragma unroll
    for (int ck = 0; ck < 2; ++ck) {
      bih[jj][ck] = *(const v16h*)(wih + n * 64 + ck * 32 + kb);
      bhh[jj][ck] = *(const v16h*)(whh + n * 64 + ck * 32 + kb);
    }
  }

  float cst[4] = {0.f, 0.f, 0.f, 0.f};       // cell state, 4 elems/thread
  float hlast[4] = {0.f, 0.f, 0.f, 0.f};     // last hidden, for final store

  // A-operand layout (16x32 f16): row M = lane&15; 16 contiguous halves at
  // K = 16*(lane>>4) .. 16*(lane>>4)+15 (matches B).
  const int mrow = lane & 15;
  const int koff = (lane >> 4) * 16;
  const f16* xrowBase = hdense + (size_t)(rb * 16 + mrow) * (LLS * CC);

  // pointwise mapping: 4 contiguous channels of one row per thread
  const int prow = (tid * 4) >> 6;
  const int pch  = (tid * 4) & 63;

  union V16 { v16h v; v8h h[2]; };

  for (int t = 0; t < LLS; ++t) {
    __syncthreads();                          // hbuf ready / gatesLds free
    V16 ah0, ah1, ax0, ax1;
    ah0.h[0] = *(const v8h*)&hbuf[mrow][koff];
    ah0.h[1] = *(const v8h*)&hbuf[mrow][8 + koff];
    ah1.h[0] = *(const v8h*)&hbuf[mrow][32 + koff];
    ah1.h[1] = *(const v8h*)&hbuf[mrow][40 + koff];
    const f16* xr = xrowBase + t * CC;
    ax0.h[0] = *(const v8h*)(xr + koff);
    ax0.h[1] = *(const v8h*)(xr + 8 + koff);
    ax1.h[0] = *(const v8h*)(xr + 32 + koff);
    ax1.h[1] = *(const v8h*)(xr + 40 + koff);

    #pragma unroll
    for (int jj = 0; jj < 2; ++jj) {
      v8f acc = {};
      acc = __builtin_amdgcn_wmma_f32_16x16x32_f16(false, ax0.v, false, bih[jj][0], (short)0, acc, false, false);
      acc = __builtin_amdgcn_wmma_f32_16x16x32_f16(false, ax1.v, false, bih[jj][1], (short)0, acc, false, false);
      acc = __builtin_amdgcn_wmma_f32_16x16x32_f16(false, ah0.v, false, bhh[jj][0], (short)0, acc, false, false);
      acc = __builtin_amdgcn_wmma_f32_16x16x32_f16(false, ah1.v, false, bhh[jj][1], (short)0, acc, false, false);
      // D layout: lanes 0-15: N=lane, M=r; lanes 16-31: N=lane-16, M=r+8
      int nCol  = (wave * 2 + jj) * 16 + (lane & 15);
      int mBase = (lane >> 4) * 8;
      #pragma unroll
      for (int r = 0; r < 8; ++r) gatesLds[mBase + r][nCol] = acc[r];
    }
    __syncthreads();                          // gates ready, hbuf consumed

    // pointwise LSTM update: vectorized 4-channel slab per thread, branchless
    v4f gi = *(const v4f*)&gatesLds[prow][pch]       + *(const v4f*)&biasLds[pch];
    v4f gf = *(const v4f*)&gatesLds[prow][64 + pch]  + *(const v4f*)&biasLds[64 + pch];
    v4f gg = *(const v4f*)&gatesLds[prow][128 + pch] + *(const v4f*)&biasLds[128 + pch];
    v4f go = *(const v4f*)&gatesLds[prow][192 + pch] + *(const v4f*)&biasLds[192 + pch];
    v4h hv4;
    #pragma unroll
    for (int i = 0; i < 4; ++i) {
      float cv = fsigm(gf[i]) * cst[i] + fsigm(gi[i]) * ftanh(gg[i]);
      cst[i] = cv;
      float hv = fsigm(go[i]) * ftanh(cv);
      hlast[i] = hv;
      hv4[i] = (f16)hv;
    }
    *(v4h*)&hbuf[prow][pch] = hv4;
  }

  // final hidden state -> output [BBG, CC] f32
  int bg = rb * 16 + prow;
  if (bg < BBG) {
    v4f o; 
    #pragma unroll
    for (int i = 0; i < 4; ++i) o[i] = hlast[i];
    *(v4f*)&out[bg * 64 + pch] = o;
  }
}

// ---------------- launcher ----------------
extern "C" void kernel_launch(void* const* d_in, const int* in_sizes, int n_in,
                              void* d_out, int out_size, void* d_ws, size_t ws_size,
                              hipStream_t stream) {
  (void)in_sizes; (void)n_in; (void)out_size; (void)ws_size;
  const float* x     = (const float*)d_in[0];
  const int*   ei    = (const int*)d_in[1];
  const float* eattr = (const float*)d_in[2];
  // d_in[3] = batch (sorted equal-size graphs -> implicit)
  const float* Wl    = (const float*)d_in[4];
  const float* bl    = (const float*)d_in[5];
  const float* Wr    = (const float*)d_in[6];
  const float* br    = (const float*)d_in[7];
  const float* We    = (const float*)d_in[8];
  const float* att   = (const float*)d_in[9];
  const float* gbias = (const float*)d_in[10];
  const float* gamma = (const float*)d_in[11];
  const float* beta  = (const float*)d_in[12];
  const float* Wih   = (const float*)d_in[13];
  const float* Whh   = (const float*)d_in[14];
  const float* bih   = (const float*)d_in[15];
  const float* bhh   = (const float*)d_in[16];
  float* out = (float*)d_out;
  char*  ws  = (char*)d_ws;

  float*    sumE   = (float*)(ws + OFF_SUME);
  unsigned* amax   = (unsigned*)(ws + OFF_AMAX);
  float*    denom  = (float*)(ws + OFF_DENOM);
  float*    T1     = (float*)(ws + OFF_T1);
  float*    bnsum  = (float*)(ws + OFF_BNSUM);
  float*    bnsq   = (float*)(ws + OFF_BNSQ);
  float*    scale  = (float*)(ws + OFF_SCALE);
  float*    shift  = (float*)(ws + OFF_SHIFT);
  float*    bias   = (float*)(ws + OFF_BIAS);
  f16*      hdense = (f16*)(ws + OFF_HDENSE);
  f16*      wihH   = (f16*)(ws + OFF_WIH);
  f16*      whhH   = (f16*)(ws + OFF_WHH);
  float*    alpha  = (float*)(ws + OFF_ALPHA);
  float*    hraw   = (float*)(ws + OFF_HRAW);

  const size_t nz = ZERO_BYTES / 16;
  k_zero<<<1024, 256, 0, stream>>>((uint4*)ws, nz);
  k_edge_sum<<<(EE + 255) / 256, 256, 0, stream>>>(eattr, sumE);
  const int egrid = (TOTE + 255) / 256;
  k_edge_alpha<<<egrid, 256, 0, stream>>>(ei, eattr, x, Wl, bl, Wr, br, We, att,
                                          sumE, alpha, amax);
  k_edge_soft<<<egrid, 256, 0, stream>>>(ei, x, alpha, amax, denom, T1);
  k_node<<<(NN * 64) / 256, 256, 0, stream>>>(denom, T1, Wl, bl, gbias, hraw, bnsum, bnsq);
  k_bnstats<<<1, 64, 0, stream>>>(bnsum, bnsq, gamma, beta, scale, shift);
  k_bnapply<<<(NN * 64) / 256, 256, 0, stream>>>(hraw, scale, shift, hdense);
  k_prep<<<(4 * CC * CC) / 256, 256, 0, stream>>>(Wih, Whh, bih, bhh, wihH, whhH, bias);
  k_lstm<<<BPAD / 16, 256, 0, stream>>>(hdense, wihH, whhH, bias, out);
}
